// GroupedQueryAttention_10445360464030
// MI455X (gfx1250) — matrix operand order repaired and checked by reading, not compiled, not run on hardware
//
#include <hip/hip_runtime.h>
#include <hip/hip_bf16.h>

// ---------------------------------------------------------------------------
// GQA attention for MI455X (gfx1250): bf16 WMMA + TDM (tensor_load_to_lds)
// double-buffered GEMM staging, flash attention with WMMA.
// ---------------------------------------------------------------------------

typedef __attribute__((ext_vector_type(16))) __bf16 v16bf;
typedef __attribute__((ext_vector_type(8)))  __bf16 v8bf;
typedef __attribute__((ext_vector_type(4)))  __bf16 v4bf;
typedef __attribute__((ext_vector_type(8)))  float  v8f;
typedef __attribute__((ext_vector_type(4)))  unsigned int v4u;
typedef __attribute__((ext_vector_type(8)))  int    v8i;
typedef __attribute__((ext_vector_type(4)))  int    v4i;

#define D_MODEL   2048
#define N_HEADS   16
#define KV_GROUPS 4
#define D_K       128
#define BATCH     2
#define SEQ       2048
#define KV_DIM    512        // KV_GROUPS * D_K

// GEMM macro tile
#define GM 128
#define GN 64
#define GK 32

// ---------------------------------------------------------------------------
// TDM: 2D tile load Global -> LDS via Tensor Data Mover descriptor (D#).
// Layout per CDNA5 ISA ch.8: group0 = {flags, lds_addr, global_addr, type=2},
// group1 = {mask/data_size/pad, tensor dims, tile dims, strides}.
// pad_interval=16 DWORDs (one 64B tile row), pad_amount=4 DWORDs (16B)
// -> LDS row stride 80B, matching the [rows][40] __bf16 arrays below.
// ---------------------------------------------------------------------------
__device__ __forceinline__ void tdm_load_2d_bf16(unsigned lds_addr,
                                                 const void* gptr,
                                                 unsigned tile_k,    // elems
                                                 unsigned tile_rows,
                                                 unsigned row_stride_elems) {
  unsigned long long ga = (unsigned long long)(size_t)gptr;
  v4u g0;
  g0[0] = 1u;                                  // count=1, user mode, no gather
  g0[1] = lds_addr;                            // LDS byte address
  g0[2] = (unsigned)(ga & 0xFFFFFFFFu);        // global_addr[31:0]
  g0[3] = (unsigned)((ga >> 32) & 0x1FFFFFFu) | (2u << 30);  // [56:32]|type=2

  const unsigned pad_interval = 3u;            // 16 DWORDs between pads
  const unsigned pad_amount   = 3u;            // 4 DWORDs of pad
  v8i g1;
  g1[0] = (int)((1u << 16)                     // data_size = 1 -> 2 bytes
              | (1u << 20)                     // pad_enable
              | (pad_interval << 22)
              | (pad_amount << 25));
  // tensor_dim0 = tile_k (no OOB), bits[79:48]
  g1[1] = (int)((tile_k & 0xFFFFu) << 16);
  // tensor_dim0[31:16]=0 | tensor_dim1[15:0] (bits[111:80])
  g1[2] = (int)((tile_rows & 0xFFFFu) << 16);
  // tensor_dim1[31:16]=0 | tile_dim0 (bits[127:112])
  g1[3] = (int)((tile_k & 0xFFFFu) << 16);
  // tile_dim1 (bits[143:128]) | tile_dim2=0
  g1[4] = (int)(tile_rows & 0xFFFFu);
  // tensor_dim0_stride[31:0] (bits[191:160])
  g1[5] = (int)row_stride_elems;
  g1[6] = 0;                                   // stride0[47:32] | stride1[15:0]
  g1[7] = 0;                                   // stride1[47:16]

  v4i z4 = {0, 0, 0, 0};
#if __has_include(<hip/amd_detail/amd_gfx1250_TDM.h>)
  v8i z8 = {0, 0, 0, 0, 0, 0, 0, 0};
  __builtin_amdgcn_tensor_load_to_lds(g0, g1, z4, z4, z8, 0);
#else
  __builtin_amdgcn_tensor_load_to_lds(g0, g1, z4, z4, 0);
#endif
}

// ---------------------------------------------------------------------------
// fp32 -> bf16 conversion (vectorized x4)
// ---------------------------------------------------------------------------
__global__ void cvt_f32_bf16_v4(const float* __restrict__ in,
                                __bf16* __restrict__ out, int n4) {
  int i = blockIdx.x * blockDim.x + threadIdx.x;
  if (i < n4) {
    float4 f = ((const float4*)in)[i];
    v4bf o;
    o[0] = (__bf16)f.x; o[1] = (__bf16)f.y;
    o[2] = (__bf16)f.z; o[3] = (__bf16)f.w;
    ((v4bf*)out)[i] = o;
  }
}

// fp32 W[k][n] -> bf16 Wt[n][k]  (one-time; makes B tiles TDM-loadable)
__global__ void cvt_transpose_f32_bf16(const float* __restrict__ in,
                                       __bf16* __restrict__ out,
                                       int K, int N) {
  int idx = blockIdx.x * blockDim.x + threadIdx.x;
  if (idx < K * N) {
    int k = idx % K;
    int n = idx / K;
    out[idx] = (__bf16)in[(size_t)k * N + n];
  }
}

// ---------------------------------------------------------------------------
// bf16 WMMA GEMM with TDM staging: C[M,N] = A[M,K] @ Bt[N,K]^T + bias
// 256 threads = 8 waves; macro tile 128(M) x 64(N); K-step 32; double-buffered
// LDS fed by tensor_load_to_lds (wave 0 issues, s_wait_tensorcnt syncs).
// Waves: 4(M) x 2(N); each wave owns a 2x2 grid of 16x16 C tiles (4 WMMAs).
// M%128==0, N%64==0, K%32==0 (true for all shapes here).
// ---------------------------------------------------------------------------
__global__ __launch_bounds__(256)
void gemm_bf16_tdm(const __bf16* __restrict__ A,   // [M,K]
                   const __bf16* __restrict__ Bt,  // [N,K]
                   const float* __restrict__ bias,
                   float* __restrict__ Cf, __bf16* __restrict__ Cbf,
                   int M, int N, int K) {
  __shared__ __bf16 As[2][GM][40];   // 128 x 32 (+8 pad) per buffer
  __shared__ __bf16 Bs[2][GN][40];   // 64  x 32 (+8 pad) per buffer

  const int t    = threadIdx.x;
  const int lane = t & 31;
  const int wave = t >> 5;
  const int half = lane >> 4;
  const int l16  = lane & 15;
  const int wm   = wave & 3;        // 0..3 -> M rows wm*32..wm*32+31
  const int wn   = wave >> 2;       // 0..1 -> N cols wn*32..wn*32+31
  const int blockM = blockIdx.y * GM;
  const int blockN = blockIdx.x * GN;

  const unsigned asBase  = (unsigned)(size_t)(void*)&As[0][0][0];
  const unsigned bsBase  = (unsigned)(size_t)(void*)&Bs[0][0][0];
  const unsigned asStage = GM * 40 * 2;   // bytes per buffer
  const unsigned bsStage = GN * 40 * 2;

  const int nstages = K / GK;

  if (wave == 0) {
    tdm_load_2d_bf16(asBase, A + (size_t)blockM * K, GK, GM, (unsigned)K);
    tdm_load_2d_bf16(bsBase, Bt + (size_t)blockN * K, GK, GN, (unsigned)K);
  }

  v8f acc[2][2];
#pragma unroll
  for (int i = 0; i < 2; ++i)
#pragma unroll
    for (int j = 0; j < 2; ++j) acc[i][j] = (v8f){};

  for (int stage = 0; stage < nstages; ++stage) {
    if (wave == 0) {
      if (stage + 1 < nstages) {
        const int nb = (stage + 1) & 1;
        tdm_load_2d_bf16(asBase + nb * asStage,
                         A + (size_t)blockM * K + (stage + 1) * GK,
                         GK, GM, (unsigned)K);
        tdm_load_2d_bf16(bsBase + nb * bsStage,
                         Bt + (size_t)blockN * K + (stage + 1) * GK,
                         GK, GN, (unsigned)K);
        __builtin_amdgcn_s_wait_tensorcnt((short)2);  // current stage done
      } else {
        __builtin_amdgcn_s_wait_tensorcnt((short)0);
      }
    }
    __syncthreads();

    const int buf = stage & 1;
    // A fragments: lane row l16, K runs {half*8..+7},{16+half*8..+7}
    v16bf af[2];
#pragma unroll
    for (int i = 0; i < 2; ++i) {
      const __bf16* ar = &As[buf][wm * 32 + i * 16 + l16][0];
      ((v8bf*)&af[i])[0] = *(const v8bf*)(ar + half * 8);
      ((v8bf*)&af[i])[1] = *(const v8bf*)(ar + 16 + half * 8);
    }
    // B fragments: lane col l16, K runs {half*8..+7},{16+half*8..+7}
    v16bf bfr[2];
#pragma unroll
    for (int j = 0; j < 2; ++j) {
      const __bf16* br = &Bs[buf][wn * 32 + j * 16 + l16][0];
      ((v8bf*)&bfr[j])[0] = *(const v8bf*)(br + half * 8);
      ((v8bf*)&bfr[j])[1] = *(const v8bf*)(br + 16 + half * 8);
    }
#pragma unroll
    for (int i = 0; i < 2; ++i)
#pragma unroll
      for (int j = 0; j < 2; ++j)
        acc[i][j] = __builtin_amdgcn_wmma_f32_16x16x32_bf16(
            false, af[i], false, bfr[j], (short)0, acc[i][j], false, false);
    __syncthreads();
  }

  // Epilogue: C layout: VGPR r -> row r + 8*half, col = l16
#pragma unroll
  for (int j = 0; j < 2; ++j) {
    const int col = blockN + wn * 32 + j * 16 + l16;
    const float bb = bias ? bias[col] : 0.0f;
#pragma unroll
    for (int i = 0; i < 2; ++i) {
#pragma unroll
      for (int r = 0; r < 8; ++r) {
        const int row = blockM + wm * 32 + i * 16 + r + 8 * half;
        const float v = acc[i][j][r] + bb;
        if (Cf)  Cf[(size_t)row * N + col]  = v;
        if (Cbf) Cbf[(size_t)row * N + col] = (__bf16)v;
      }
    }
  }
}

// ---------------------------------------------------------------------------
// V transpose: V[b,s,g,d] (bf16, [B*S,512]) -> Vt[b,g,d,t] ([B*G*128, S])
// ---------------------------------------------------------------------------
__global__ void transpose_v(const __bf16* __restrict__ V,
                            __bf16* __restrict__ Vt) {
  int idx = blockIdx.x * blockDim.x + threadIdx.x;   // 0 .. 2^21-1
  int t = idx & (SEQ - 1);
  int d = (idx >> 11) & 127;
  int g = (idx >> 18) & 3;
  int b = idx >> 20;
  Vt[idx] = V[(size_t)(b * SEQ + t) * KV_DIM + g * D_K + d];
}

// ---------------------------------------------------------------------------
// Flash attention: 1 wave per (b, head, 16-query tile).
// ---------------------------------------------------------------------------
__global__ __launch_bounds__(32)
void gqa_flash_attn(const __bf16* __restrict__ Q,   // [B*S, 2048]
                    const __bf16* __restrict__ Kb,  // [B*S, 512]
                    const __bf16* __restrict__ Vt,  // [B*G*128, S]
                    __bf16* __restrict__ O) {       // [B*S, 2048]
  __shared__ __bf16 Plds[16][40];

  const int lane = threadIdx.x & 31;
  const int half = lane >> 4;
  const int l16  = lane & 15;
  const int qt   = blockIdx.x;
  const int head = blockIdx.y;
  const int b    = blockIdx.z;
  const int g    = head >> 2;

  const float scale = 0.08838834764831845f;  // 1/sqrt(128)

  v16bf qa[4];
  {
    const __bf16* qrow =
        Q + (size_t)(b * SEQ + qt * 16 + l16) * D_MODEL + head * D_K;
#pragma unroll
    for (int ks = 0; ks < 4; ++ks) {
      ((v8bf*)&qa[ks])[0] = *(const v8bf*)(qrow + ks * 32 + half * 8);
      ((v8bf*)&qa[ks])[1] = *(const v8bf*)(qrow + ks * 32 + 16 + half * 8);
    }
  }

  v8f o_acc[8];
#pragma unroll
  for (int d = 0; d < 8; ++d) o_acc[d] = (v8f){};
  float m_r[8], l_r[8];
#pragma unroll
  for (int r = 0; r < 8; ++r) { m_r[r] = -1e30f; l_r[r] = 0.0f; }

  for (int kb = 0; kb < SEQ / 32; ++kb) {
    const int t0 = kb * 32;
    v8f s0 = {}, s1 = {};
    {
      const __bf16* kr0 =
          Kb + (size_t)(b * SEQ + t0 + l16) * KV_DIM + g * D_K;
      const __bf16* kr1 = kr0 + (size_t)16 * KV_DIM;
      if (kb + 1 < SEQ / 32)   // prefetch next chunk's K rows
        __builtin_prefetch(kr0 + (size_t)32 * KV_DIM, 0, 1);
#pragma unroll
      for (int ks = 0; ks < 4; ++ks) {
        v16bf kf0, kf1;
        ((v8bf*)&kf0)[0] = *(const v8bf*)(kr0 + ks * 32 + half * 8);
        ((v8bf*)&kf0)[1] = *(const v8bf*)(kr0 + ks * 32 + 16 + half * 8);
        ((v8bf*)&kf1)[0] = *(const v8bf*)(kr1 + ks * 32 + half * 8);
        ((v8bf*)&kf1)[1] = *(const v8bf*)(kr1 + ks * 32 + 16 + half * 8);
        s0 = __builtin_amdgcn_wmma_f32_16x16x32_bf16(false, qa[ks], false, kf0,
                                                     (short)0, s0, false, false);
        s1 = __builtin_amdgcn_wmma_f32_16x16x32_bf16(false, qa[ks], false, kf1,
                                                     (short)0, s1, false, false);
      }
    }

    float alpha[8];
#pragma unroll
    for (int r = 0; r < 8; ++r) {
      float a = s0[r] * scale;
      float c = s1[r] * scale;
      float mx = fmaxf(a, c);
#pragma unroll
      for (int mk = 1; mk < 16; mk <<= 1) mx = fmaxf(mx, __shfl_xor(mx, mk, 32));
      const float nm = fmaxf(m_r[r], mx);
      const float al = __expf(m_r[r] - nm);
      const float p0 = __expf(a - nm);
      const float p1 = __expf(c - nm);
      float rs = p0 + p1;
#pragma unroll
      for (int mk = 1; mk < 16; mk <<= 1) rs += __shfl_xor(rs, mk, 32);
      l_r[r] = l_r[r] * al + rs;
      m_r[r] = nm;
      alpha[r] = al;
      Plds[r + 8 * half][l16]      = (__bf16)p0;
      Plds[r + 8 * half][16 + l16] = (__bf16)p1;
    }
    __syncthreads();

    v16bf pf;
    ((v8bf*)&pf)[0] = *(const v8bf*)(&Plds[l16][half * 8]);
    ((v8bf*)&pf)[1] = *(const v8bf*)(&Plds[l16][16 + half * 8]);

#pragma unroll
    for (int d = 0; d < 8; ++d)
#pragma unroll
      for (int r = 0; r < 8; ++r) o_acc[d][r] *= alpha[r];

    const __bf16* vbase =
        Vt + (size_t)((b * KV_GROUPS + g) * D_K) * SEQ + t0;
    if (kb + 1 < SEQ / 32)
      __builtin_prefetch(vbase + (size_t)l16 * SEQ + 32, 0, 1);
#pragma unroll
    for (int d = 0; d < 8; ++d) {
      const __bf16* vrow = vbase + (size_t)(d * 16 + l16) * SEQ + half * 8;
      v16bf vf;
      ((v8bf*)&vf)[0] = *(const v8bf*)(vrow);
      ((v8bf*)&vf)[1] = *(const v8bf*)(vrow + 16);
      o_acc[d] = __builtin_amdgcn_wmma_f32_16x16x32_bf16(false, pf, false, vf,
                                                         (short)0, o_acc[d],
                                                         false, false);
    }
    __syncthreads();
  }

#pragma unroll
  for (int r = 0; r < 8; ++r) {
    const float inv = 1.0f / l_r[r];
    __bf16* orow = O + (size_t)(b * SEQ + qt * 16 + r + 8 * half) * D_MODEL +
                   head * D_K + l16;
#pragma unroll
    for (int d = 0; d < 8; ++d) orow[d * 16] = (__bf16)(o_acc[d][r] * inv);
  }
}

// ---------------------------------------------------------------------------
// Host-side orchestration
// ---------------------------------------------------------------------------
extern "C" void kernel_launch(void* const* d_in, const int* in_sizes, int n_in,
                              void* d_out, int out_size, void* d_ws,
                              size_t ws_size, hipStream_t stream) {
  const float* x  = (const float*)d_in[0];
  const float* Wq = (const float*)d_in[1];
  const float* bq = (const float*)d_in[2];
  const float* Wk = (const float*)d_in[3];
  const float* bk = (const float*)d_in[4];
  const float* Wv = (const float*)d_in[5];
  const float* bv = (const float*)d_in[6];
  const float* Wo = (const float*)d_in[7];
  const float* bo = (const float*)d_in[8];
  float* out = (float*)d_out;

  const size_t MS = (size_t)BATCH * SEQ;  // 4096 token rows

  char* ws = (char*)d_ws;
  size_t off = 0;
  auto alloc = [&](size_t bytes) -> void* {
    void* p = ws + off;
    off += (bytes + 255) & ~(size_t)255;
    return p;
  };
  __bf16* xbf  = (__bf16*)alloc(MS * D_MODEL * 2);
  __bf16* wqT  = (__bf16*)alloc((size_t)D_MODEL * D_MODEL * 2);  // [N=2048][K]
  __bf16* wkT  = (__bf16*)alloc((size_t)KV_DIM * D_MODEL * 2);   // [N=512][K]
  __bf16* wvT  = (__bf16*)alloc((size_t)KV_DIM * D_MODEL * 2);
  __bf16* woT  = (__bf16*)alloc((size_t)D_MODEL * D_MODEL * 2);
  __bf16* Qbf  = (__bf16*)alloc(MS * D_MODEL * 2);
  __bf16* Kbf  = (__bf16*)alloc(MS * KV_DIM * 2);
  __bf16* Vbf  = (__bf16*)alloc(MS * KV_DIM * 2);
  __bf16* Vtbf = (__bf16*)alloc(MS * KV_DIM * 2);
  __bf16* Obf  = (__bf16*)alloc(MS * D_MODEL * 2);

  {
    int n4 = (int)(MS * D_MODEL / 4);
    cvt_f32_bf16_v4<<<(n4 + 255) / 256, 256, 0, stream>>>(x, xbf, n4);
  }
  auto cvtT = [&](const float* src, __bf16* dst, int K, int N) {
    int n = K * N;
    cvt_transpose_f32_bf16<<<(n + 255) / 256, 256, 0, stream>>>(src, dst, K, N);
  };
  cvtT(Wq, wqT, D_MODEL, D_MODEL);
  cvtT(Wk, wkT, D_MODEL, KV_DIM);
  cvtT(Wv, wvT, D_MODEL, KV_DIM);
  cvtT(Wo, woT, D_MODEL, D_MODEL);

  // Projections (bf16 out): C = x @ W + b, with W given transposed [N][K].
  gemm_bf16_tdm<<<dim3(D_MODEL / GN, MS / GM), 256, 0, stream>>>(
      xbf, wqT, bq, nullptr, Qbf, (int)MS, D_MODEL, D_MODEL);
  gemm_bf16_tdm<<<dim3(KV_DIM / GN, MS / GM), 256, 0, stream>>>(
      xbf, wkT, bk, nullptr, Kbf, (int)MS, KV_DIM, D_MODEL);
  gemm_bf16_tdm<<<dim3(KV_DIM / GN, MS / GM), 256, 0, stream>>>(
      xbf, wvT, bv, nullptr, Vbf, (int)MS, KV_DIM, D_MODEL);

  {
    int n = BATCH * KV_GROUPS * D_K * SEQ;  // 2^21
    transpose_v<<<n / 256, 256, 0, stream>>>(Vbf, Vtbf);
  }

  gqa_flash_attn<<<dim3(SEQ / 16, N_HEADS, BATCH), 32, 0, stream>>>(
      Qbf, Kbf, Vtbf, Obf);

  // Output projection (f32 out): out = O @ Wo + bo.
  gemm_bf16_tdm<<<dim3(D_MODEL / GN, MS / GM), 256, 0, stream>>>(
      Obf, woT, bo, out, nullptr, (int)MS, D_MODEL, D_MODEL);
}
